// PixelAttention_45698452030346
// MI455X (gfx1250) — matrix operand order repaired and checked by reading, not compiled, not run on hardware
//
#include <hip/hip_runtime.h>
#include <hip/hip_bf16.h>

// ---------------------------------------------------------------------------
// MI455X (gfx1250): bf16 WMMA GEMMs with fragment-order-matched memory layouts
// (weights pre-permuted to fragment element order; LDS tiles transposed to
// ci-innermost) so A/B fragments load as b128 vectors instead of 16 scalar
// gathers. Async global->LDS (ASYNCcnt) stages the stage-1 input tile.
// ---------------------------------------------------------------------------

typedef __attribute__((ext_vector_type(16))) __bf16 v16bf;
typedef __attribute__((ext_vector_type(8)))  __bf16 v8bf;   // 16 B
typedef __attribute__((ext_vector_type(8)))  float  v8f;

#define HW  65536   // 256*256
#define IMG 256

// WMMA 16x16x32 bf16 A-fragment k-index (contiguous, matching B): element i of lane
// (m = lane&15, half = lane>>4) holds k = ak_of(i, half) = i + 16*half.
__device__ __forceinline__ int ak_of(int i, int half) {
  return i + 16 * half;
}
// B-fragment: element i holds k = i + 16*half (contiguous per half).

__device__ __forceinline__ v16bf load_v16bf(const __bf16* p) {   // 2 x 16B chunks
  v8bf lo = *(const v8bf*)(p);
  v8bf hi = *(const v8bf*)(p + 8);
  return __builtin_shufflevector(lo, hi, 0, 1, 2, 3, 4, 5, 6, 7,
                                 8, 9, 10, 11, 12, 13, 14, 15);
}
__device__ __forceinline__ v16bf cvt_row(const float* p) {       // 16 f32 -> v16bf
  v16bf r;
#pragma unroll
  for (int i = 0; i < 16; i++) r[i] = (__bf16)p[i];
  return r;
}

__device__ __forceinline__ float bf2f(unsigned short u) {
  return __builtin_bit_cast(float, (unsigned int)u << 16);
}
__device__ __forceinline__ void unpack8(uint4 v, float* f) {
  unsigned int u[4] = {v.x, v.y, v.z, v.w};
#pragma unroll
  for (int i = 0; i < 4; i++) {
    f[2 * i + 0] = bf2f((unsigned short)(u[i] & 0xffffu));
    f[2 * i + 1] = bf2f((unsigned short)(u[i] >> 16));
  }
}

// gfx1250 async copy: one fp32 element global -> LDS, tracked by ASYNCcnt.
__device__ __forceinline__ void async_ld_f32(unsigned lds_addr, int byte_off,
                                             const float* sbase) {
  asm volatile("global_load_async_to_lds_b32 %0, %1, %2"
               :: "v"(lds_addr), "v"(byte_off), "s"(sbase)
               : "memory");
}
__device__ __forceinline__ void wait_async0() {
  asm volatile("s_wait_asynccnt 0x0" ::: "memory");
}

// ---------------------------------------------------------------------------
// Weight repack kernels (run once): f32 -> bf16, rows permuted into WMMA
// A-fragment element order (per 32-ci block, pos j = 16*half+i <-> ak_of(i,half)).
// ---------------------------------------------------------------------------
__global__ void repack_1x1_kernel(const float* __restrict__ src, __bf16* __restrict__ dst,
                                  int nrows) {   // src row-major [row][32]
  const int idx = blockIdx.x * 256 + threadIdx.x;
  if (idx >= nrows * 32) return;
  const int row = idx >> 5, j = idx & 31;
  const int ci = ak_of(j & 15, j >> 4);
  dst[idx] = (__bf16)src[row * 32 + ci];
}

__global__ void repack_conv_kernel(const float* __restrict__ src, __bf16* __restrict__ dst,
                                   int Cout, int Cin) { // src OIHW; dst [tap][co][ci(perm)]
  const int idx = blockIdx.x * 256 + threadIdx.x;
  if (idx >= 9 * Cout * Cin) return;
  const int cip = idx % Cin;
  const int rest = idx / Cin;
  const int co = rest % Cout, tap = rest / Cout;
  const int j = cip & 31;
  const int src_ci = (cip & ~31) + ak_of(j & 15, j >> 4);
  dst[idx] = (__bf16)src[(co * Cin + src_ci) * 9 + tap];
}

// ---------------------------------------------------------------------------
// Stage 1: 1x1 conv QKV, shifts baked into K/V outputs (zero-padded OOB).
// grid = (256 rows, 19 groups, 4 batches), 256 threads (8 waves).
// LDS tile is [col][ci] fp32 (stride 36 floats: 16B-aligned, bank-rotating),
// staged with async global->LDS; B fragments are contiguous ds_load_b128 runs.
// ---------------------------------------------------------------------------
__global__ void qkv_kernel(const float* __restrict__ x,
                           const __bf16* __restrict__ wkT, const float* __restrict__ bk,
                           const __bf16* __restrict__ wvT, const float* __restrict__ bv,
                           const __bf16* __restrict__ wqT, const float* __restrict__ bq,
                           __bf16* __restrict__ Kimg, __bf16* __restrict__ Vimg,
                           __bf16* __restrict__ Qimg) {
  __shared__ float xs[IMG][36];   // [col][ci], 36 KB

  const int h = blockIdx.x;
  const int g = blockIdx.y;
  const int b = blockIdx.z;

  const __bf16* Wt;
  const float* bias;
  __bf16* dst;
  int di = 0, dj = 0;
  if (g < 9) {
    int p = g;
    Wt = wkT + (size_t)p * 1024; bias = bk + p * 32;
    di = p / 3 - 1; dj = p % 3 - 1;
    dst = Kimg + ((size_t)b * 288 + (size_t)p * 32) * HW;
  } else if (g < 18) {
    int p = g - 9;
    Wt = wvT + (size_t)p * 1024; bias = bv + p * 32;
    di = p / 3 - 1; dj = p % 3 - 1;
    dst = Vimg + ((size_t)b * 288 + (size_t)p * 32) * HW;
  } else {
    Wt = wqT; bias = bq;
    dst = Qimg + (size_t)b * 32 * HW;
  }

  const int hh = h + di;
  const bool rowOK = (unsigned)hh < (unsigned)IMG;  // block-uniform
  const int tid = threadIdx.x;
  const float* xb = x + (((size_t)b * 32) << 16);   // uniform batch base

  const unsigned sbase = (unsigned)(uintptr_t)(&xs[0][0]);
  if (rowOK) {
    for (int idx = tid; idx < 32 * IMG; idx += 256) {
      const int ci = idx >> 8, wcol = idx & 255;    // consecutive tid -> consecutive ww
      const int ww = wcol + dj;
      if ((unsigned)ww < (unsigned)IMG) {
        const unsigned lds_addr = sbase + (unsigned)(wcol * 36 + ci) * 4u;
        const int goff = ((ci << 16) + hh * IMG + ww) << 2;
        async_ld_f32(lds_addr, goff, xb);      // global_load_async_to_lds_b32
      } else {
        xs[wcol][ci] = 0.0f;
      }
    }
    wait_async0();                              // s_wait_asynccnt 0
  } else {
    for (int idx = tid; idx < 32 * IMG; idx += 256)
      xs[idx & 255][idx >> 8] = 0.0f;
  }
  __syncthreads();

  const int lane = tid & 31, wave = tid >> 5;
  const int half = lane >> 4, ln = lane & 15;
  const int w0 = wave * 32;

  // A: one contiguous 32B run per fragment (weights pre-permuted).
  v16bf a0 = load_v16bf(wkT == Wt || true ? Wt + (size_t)ln * 32 + 16 * half
                                          : Wt);   // (kept simple; same expr)
  v16bf a1 = load_v16bf(Wt + (size_t)(16 + ln) * 32 + 16 * half);
  // B: contiguous 16-f32 run in LDS, converted to bf16.
  v16bf b0 = cvt_row(&xs[w0 + ln][16 * half]);
  v16bf b1 = cvt_row(&xs[w0 + 16 + ln][16 * half]);

  v8f z = {};
  v8f d[2][2];
  d[0][0] = __builtin_amdgcn_wmma_f32_16x16x32_bf16(false, a0, false, b0, (short)0, z, false, false);
  d[0][1] = __builtin_amdgcn_wmma_f32_16x16x32_bf16(false, a0, false, b1, (short)0, z, false, false);
  d[1][0] = __builtin_amdgcn_wmma_f32_16x16x32_bf16(false, a1, false, b0, (short)0, z, false, false);
  d[1][1] = __builtin_amdgcn_wmma_f32_16x16x32_bf16(false, a1, false, b1, (short)0, z, false, false);

  float bia[2][8];
#pragma unroll
  for (int ct = 0; ct < 2; ct++)
#pragma unroll
    for (int v = 0; v < 8; v++)
      bia[ct][v] = bias[ct * 16 + v + 8 * half];

#pragma unroll
  for (int ct = 0; ct < 2; ct++)
#pragma unroll
    for (int pt = 0; pt < 2; pt++)
#pragma unroll
      for (int v = 0; v < 8; v++) {
        const int co = ct * 16 + v + 8 * half;           // D layout: lanes16-31 -> M+8
        const int px = w0 + pt * 16 + ln;
        const int ww = px + dj;
        const bool ok = rowOK && (unsigned)ww < (unsigned)IMG;
        const float val = ok ? d[ct][pt][v] + bia[ct][v] : 0.0f;
        dst[((size_t)co << 16) + h * IMG + px] = (__bf16)val;
      }
}

// ---------------------------------------------------------------------------
// Stage 2: scrambled-view attention. One thread per flat row n (B*HW total).
// ---------------------------------------------------------------------------
__global__ void attn_kernel(const __bf16* __restrict__ K, const __bf16* __restrict__ V,
                            const __bf16* __restrict__ Q, float* __restrict__ t0) {
  const int n = blockIdx.x * blockDim.x + threadIdx.x;
  const int b = n >> 16, nl = n & 0xffff;
  const __bf16* Kb = K + (size_t)b * 288 * HW;
  const __bf16* Vb = V + (size_t)b * 288 * HW;
  const __bf16* Qb = Q + (size_t)b * 32 * HW;

  float q[32];
  {
    const uint4* qp = (const uint4*)(Qb + (size_t)nl * 32);
#pragma unroll
    for (int i = 0; i < 4; i++) unpack8(qp[i], q + 8 * i);
  }

  const size_t f0 = (size_t)nl * 288;
  __builtin_prefetch(Vb + f0, 0, 3);   // global_prefetch_b8 for the V row

  float s[9];
#pragma unroll
  for (int p = 0; p < 9; p++) {
    const uint4* kp = (const uint4*)(Kb + f0 + p * 32);
    float dot = 0.0f;
#pragma unroll
    for (int i = 0; i < 4; i++) {
      float kf[8];
      unpack8(kp[i], kf);
#pragma unroll
      for (int j = 0; j < 8; j++) dot += kf[j] * q[8 * i + j];
    }
    const int gidx = nl * 9 + p;                 // mask's own flat scramble
    const int pm = gidx >> 16, hwm = gidx & 0xffff;
    const int hm = (hwm >> 8) + (pm / 3 - 1);
    const int wm = (hwm & 255) + (pm % 3 - 1);
    const bool ok = ((unsigned)hm < (unsigned)IMG) && ((unsigned)wm < (unsigned)IMG);
    s[p] = ok ? dot * 0.125f : 0.0f;
  }

  float mx = s[0];
#pragma unroll
  for (int p = 1; p < 9; p++) mx = fmaxf(mx, s[p]);
  float e[9], sum = 0.0f;
#pragma unroll
  for (int p = 0; p < 9; p++) { e[p] = __expf(s[p] - mx); sum += e[p]; }
  const float inv = 1.0f / sum;

  float acc[32];
#pragma unroll
  for (int i = 0; i < 32; i++) acc[i] = 0.0f;
#pragma unroll
  for (int p = 0; p < 9; p++) {
    const float al = e[p] * inv;
    const uint4* vp = (const uint4*)(Vb + f0 + p * 32);
#pragma unroll
    for (int i = 0; i < 4; i++) {
      float vf[8];
      unpack8(vp[i], vf);
#pragma unroll
      for (int j = 0; j < 8; j++) acc[8 * i + j] += al * vf[j];
    }
  }

  float4* op = (float4*)(t0 + ((size_t)b << 21) + (size_t)nl * 32);
#pragma unroll
  for (int i = 0; i < 8; i++)
    op[i] = make_float4(acc[4 * i], acc[4 * i + 1], acc[4 * i + 2], acc[4 * i + 3]);
}

// ---------------------------------------------------------------------------
// Stage 3: 3x3 conv + bias + relu, implicit GEMM. grid=(256, CoutT/32, 4).
// LDS tile [kh][col][ci] bf16 (stride 40: 16B-aligned); B fragment = 2x
// ds_load_b128; A fragment = contiguous 32B from pre-permuted bf16 weights.
// ---------------------------------------------------------------------------
__global__ void conv3x3_kernel(const float* __restrict__ in, const __bf16* __restrict__ wT,
                               const float* __restrict__ bias, float* __restrict__ out,
                               int Cin, int CoutT) {
  __shared__ __bf16 xs[3][264][40];   // [kh][col][ci], ~63.4 KB

  const int h = blockIdx.x;
  const int cob = blockIdx.y * 32;
  const int b = blockIdx.z;
  const int tid = threadIdx.x, lane = tid & 31, wave = tid >> 5;
  const int half = lane >> 4, ln = lane & 15;
  const int w0 = wave * 32;

  v8f zero = {};
  v8f acc[2][2];
  acc[0][0] = zero; acc[0][1] = zero; acc[1][0] = zero; acc[1][1] = zero;

  const int nblk = Cin >> 5;
  for (int cb = 0; cb < nblk; cb++) {
    __syncthreads();
    for (int idx = tid; idx < 32 * 3 * 258; idx += 256) {
      const int ci = idx / 774, rem = idx % 774;   // consecutive tid -> consecutive col
      const int r = rem / 258, c = rem % 258;
      const int hh = h + r - 1, ww = c - 1;
      float v = 0.0f;
      if ((unsigned)hh < (unsigned)IMG && (unsigned)ww < (unsigned)IMG)
        v = in[(((size_t)b * Cin + cb * 32 + ci) << 16) + hh * IMG + ww];
      xs[r][c][ci] = (__bf16)v;
    }
    __syncthreads();

#pragma unroll
    for (int kh = 0; kh < 3; kh++)
#pragma unroll
      for (int kw = 0; kw < 3; kw++) {
        const __bf16* arow =
            wT + ((size_t)(kh * 3 + kw) * CoutT + cob + ln) * Cin + cb * 32 + 16 * half;
        v16bf a0 = load_v16bf(arow);
        v16bf a1 = load_v16bf(arow + (size_t)16 * Cin);
        v16bf b0 = load_v16bf(&xs[kh][w0 + ln + kw][16 * half]);
        v16bf b1 = load_v16bf(&xs[kh][w0 + 16 + ln + kw][16 * half]);
        acc[0][0] = __builtin_amdgcn_wmma_f32_16x16x32_bf16(false, a0, false, b0, (short)0, acc[0][0], false, false);
        acc[0][1] = __builtin_amdgcn_wmma_f32_16x16x32_bf16(false, a0, false, b1, (short)0, acc[0][1], false, false);
        acc[1][0] = __builtin_amdgcn_wmma_f32_16x16x32_bf16(false, a1, false, b0, (short)0, acc[1][0], false, false);
        acc[1][1] = __builtin_amdgcn_wmma_f32_16x16x32_bf16(false, a1, false, b1, (short)0, acc[1][1], false, false);
      }
  }

#pragma unroll
  for (int ct = 0; ct < 2; ct++)
#pragma unroll
    for (int pt = 0; pt < 2; pt++)
#pragma unroll
      for (int v = 0; v < 8; v++) {
        const int co = cob + ct * 16 + v + 8 * half;
        const int px = w0 + pt * 16 + ln;
        const float val = fmaxf(acc[ct][pt][v] + bias[co], 0.0f);
        out[(((size_t)b * CoutT + co) << 16) + h * IMG + px] = val;
      }
}

// ---------------------------------------------------------------------------
extern "C" void kernel_launch(void* const* d_in, const int* in_sizes, int n_in,
                              void* d_out, int out_size, void* d_ws, size_t ws_size,
                              hipStream_t stream) {
  (void)in_sizes; (void)n_in; (void)out_size; (void)ws_size;
  const float* x  = (const float*)d_in[0];
  const float* wk = (const float*)d_in[1];
  const float* bk = (const float*)d_in[2];
  const float* wv = (const float*)d_in[3];
  const float* bv = (const float*)d_in[4];
  const float* wq = (const float*)d_in[5];
  const float* bq = (const float*)d_in[6];
  const float* w1 = (const float*)d_in[7];
  const float* b1 = (const float*)d_in[8];
  const float* w2 = (const float*)d_in[9];
  const float* b2 = (const float*)d_in[10];
  const float* w3 = (const float*)d_in[11];
  const float* b3 = (const float*)d_in[12];
  float* outp = (float*)d_out;

  char* ws = (char*)d_ws;
  const size_t KV_BYTES = (size_t)4 * 288 * HW * 2;  // 150,994,944 (bf16)
  const size_t Q_BYTES  = (size_t)4 * 32 * HW * 2;   //  16,777,216 (bf16)
  const size_t T32      = (size_t)4 * 32 * HW * 4;   //  33,554,432 (f32)
  size_t off = 0;
  __bf16* Kimg = (__bf16*)(ws + off); off += KV_BYTES;
  __bf16* Vimg = (__bf16*)(ws + off); off += KV_BYTES;
  __bf16* Qimg = (__bf16*)(ws + off); off += Q_BYTES;
  float*  t0   = (float*) (ws + off); off += T32;
  float*  t1   = (float*) (ws + off); off += T32;
  float*  t2   = (float*) (ws + off); off += 2 * T32;  // 64-channel tensor
  __bf16* wkT  = (__bf16*)(ws + off); off += 9216 * 2;
  __bf16* wvT  = (__bf16*)(ws + off); off += 9216 * 2;
  __bf16* wqT  = (__bf16*)(ws + off); off += 1024 * 2;
  __bf16* wT1  = (__bf16*)(ws + off); off += 9216 * 2;
  __bf16* wT2  = (__bf16*)(ws + off); off += 18432 * 2;
  __bf16* wT3  = (__bf16*)(ws + off); off += 18432 * 2;

  // Stage 0: one-time weight repack into WMMA fragment order (bf16).
  repack_1x1_kernel<<<dim3(36), 256, 0, stream>>>(wk, wkT, 288);
  repack_1x1_kernel<<<dim3(36), 256, 0, stream>>>(wv, wvT, 288);
  repack_1x1_kernel<<<dim3(4),  256, 0, stream>>>(wq, wqT, 32);
  repack_conv_kernel<<<dim3(36), 256, 0, stream>>>(w1, wT1, 32, 32);
  repack_conv_kernel<<<dim3(72), 256, 0, stream>>>(w2, wT2, 64, 32);
  repack_conv_kernel<<<dim3(72), 256, 0, stream>>>(w3, wT3, 32, 64);

  // Stage 1: QKV 1x1 convs with shift/pad baked in (async global->LDS staging).
  qkv_kernel<<<dim3(IMG, 19, 4), 256, 0, stream>>>(x, wkT, bk, wvT, bv, wqT, bq,
                                                   Kimg, Vimg, Qimg);
  // Stage 2: scrambled-view attention.
  attn_kernel<<<dim3(1024), 256, 0, stream>>>(Kimg, Vimg, Qimg, t0);
  // Stage 3: three 3x3 conv+relu layers.
  conv3x3_kernel<<<dim3(IMG, 1, 4), 256, 0, stream>>>(t0, wT1, b1, t1, 32, 32);
  conv3x3_kernel<<<dim3(IMG, 2, 4), 256, 0, stream>>>(t1, wT2, b2, t2, 32, 64);
  conv3x3_kernel<<<dim3(IMG, 1, 4), 256, 0, stream>>>(t2, wT3, b3, outp, 64, 32);
}
